// FullyContextualEmbedding_83958020702527
// MI455X (gfx1250) — matrix operand order repaired and checked by reading, not compiled, not run on hardware
//
#include <hip/hip_runtime.h>
#include <hip/hip_bf16.h>

// ---------------------------------------------------------------------------
// FullyContextualEmbedding: 64-step attention-LSTM, B=1024, F=1024, S=64.
// bf16 WMMA for all GEMMs, f32 accumulate. query@W_ih_left^T + biases hoisted
// into G0 (constant), which seeds the per-step gate GEMM accumulator.
// Per-step gate GEMM fused to a single K=2048 pass via Wcat=[W_ih_right|W_hh].
// K-loops are software-pipelined (double-buffered fragments) so WMMAs overlap
// with the next chunk's global loads instead of s_wait_loadcnt 0.
// ---------------------------------------------------------------------------

typedef __attribute__((ext_vector_type(16))) __bf16 bf16x16;
typedef __attribute__((ext_vector_type(8)))  float  v8f;

#define Bsz  1024
#define FEAT 1024
#define SS   64

__device__ __forceinline__ unsigned short f2bf(float x) {
  unsigned int u = __float_as_uint(x);
  u += 0x7FFFu + ((u >> 16) & 1u);           // round-to-nearest-even
  return (unsigned short)(u >> 16);
}
__device__ __forceinline__ float fast_sig(float x) {
  return 1.0f / (1.0f + __expf(-x));
}
__device__ __forceinline__ float fast_tanh(float x) {
  return 2.0f * fast_sig(2.0f * x) - 1.0f;
}

// 16-element bf16 fragment = two 16B chunks (second at +`sec` elements)
__device__ __forceinline__ bf16x16 load_frag(const unsigned short* p, int sec) {
  union { uint4 q[2]; bf16x16 v; } u;
  u.q[0] = *reinterpret_cast<const uint4*>(p);
  u.q[1] = *reinterpret_cast<const uint4*>(p + sec);
  return u.v;
}

// ---------------------------------------------------------------------------
// WMMA GEMM:  C[M,N] = (INIT ? Cinit : 0) + A[M,K] * W[N,K]^T  (bf16 in)
// Block = 256 thr (8 waves). Wave tile 64x64 (4x4 of 16x16).
// Block tile 128x256. grid = (N/256, M/128). OUTBF: emit bf16, else f32.
// ---------------------------------------------------------------------------
template<int INIT, int OUTBF>
__global__ __launch_bounds__(256) void wmma_gemm(
    const unsigned short* __restrict__ A, int lda,
    const unsigned short* __restrict__ W, int ldb,
    const float* __restrict__ Cinit,
    void* __restrict__ Cv, int ldc, int K)
{
  const int lane  = threadIdx.x & 31;
  const int wv    = threadIdx.x >> 5;
  const int waveM = wv >> 2;                 // 0..1
  const int waveN = wv & 3;                  // 0..3
  const int rowBase = blockIdx.y * 128 + waveM * 64;
  const int colBase = blockIdx.x * 256 + waveN * 64;
  const int lh  = lane >> 4;                 // half-wave id
  const int l15 = lane & 15;

  const unsigned short* Abase = A + (size_t)(rowBase + l15) * lda + lh * 8;
  const unsigned short* Wbase = W + (size_t)(colBase + l15) * ldb + lh * 8;
  const size_t aRowStep = (size_t)16 * lda;
  const size_t bColStep = (size_t)16 * ldb;

  v8f acc[4][4];
  if (INIT) {
#pragma unroll
    for (int tm = 0; tm < 4; ++tm)
#pragma unroll
      for (int tn = 0; tn < 4; ++tn) {
        const int r0 = rowBase + tm * 16 + lh * 8;
        const int c0 = colBase + tn * 16 + l15;
#pragma unroll
        for (int r = 0; r < 8; ++r)
          acc[tm][tn][r] = Cinit[(size_t)(r0 + r) * ldc + c0];
      }
  } else {
#pragma unroll
    for (int tm = 0; tm < 4; ++tm)
#pragma unroll
      for (int tn = 0; tn < 4; ++tn)
#pragma unroll
        for (int r = 0; r < 8; ++r)
          acc[tm][tn][r] = 0.0f;
  }

  bf16x16 a0[4], b0[4], a1[4], b1[4];
  auto loadA = [&](bf16x16* aF, int k) {
#pragma unroll
    for (int tm = 0; tm < 4; ++tm)
      aF[tm] = load_frag(Abase + tm * aRowStep + k, 16);
  };
  auto loadB = [&](bf16x16* bF, int k) {
#pragma unroll
    for (int tn = 0; tn < 4; ++tn)
      bF[tn] = load_frag(Wbase + tn * bColStep + k, 16);
  };
  auto mma = [&](const bf16x16* aF, const bf16x16* bF) {
#pragma unroll
    for (int tm = 0; tm < 4; ++tm)
#pragma unroll
      for (int tn = 0; tn < 4; ++tn)
        acc[tm][tn] = __builtin_amdgcn_wmma_f32_16x16x32_bf16(
            false, aF[tm], false, bF[tn], (short)0, acc[tm][tn], false, false);
  };

  // Software pipeline over K chunks of 32 (K/32 is even for all call sites).
  loadA(a0, 0); loadB(b0, 0);
  int k = 32;
  for (; k + 32 < K; k += 64) {
    loadA(a1, k);      loadB(b1, k);        // prefetch chunk k/32
    mma(a0, b0);                            // compute chunk k/32-1
    loadA(a0, k + 32); loadB(b0, k + 32);   // prefetch chunk k/32+1
    mma(a1, b1);                            // compute chunk k/32
  }
  loadA(a1, K - 32); loadB(b1, K - 32);     // last chunk
  mma(a0, b0);
  mma(a1, b1);

#pragma unroll
  for (int tm = 0; tm < 4; ++tm)
#pragma unroll
    for (int tn = 0; tn < 4; ++tn) {
      const int r0 = rowBase + tm * 16 + lh * 8;
      const int c0 = colBase + tn * 16 + l15;
      if (OUTBF) {
        unsigned short* Cb = (unsigned short*)Cv;
#pragma unroll
        for (int r = 0; r < 8; ++r)
          Cb[(size_t)(r0 + r) * ldc + c0] = f2bf(acc[tm][tn][r]);
      } else {
        float* Cf = (float*)Cv;
#pragma unroll
        for (int r = 0; r < 8; ++r)
          Cf[(size_t)(r0 + r) * ldc + c0] = acc[tm][tn][r];
      }
    }
}

// ---------------------------------------------------------------------------
// Attention logits + fused row softmax -> bf16 attn[B,64].
// Wave computes 16 rows x 64 cols (1x4 tiles), K = FEAT, pipelined.
// grid.x = B/128, block = 256.
// ---------------------------------------------------------------------------
__global__ __launch_bounds__(256) void attn_softmax_kernel(
    const unsigned short* __restrict__ H, int lda,       // h bf16 [B,*]
    const unsigned short* __restrict__ Wsup,             // support bf16 [64,FEAT]
    unsigned short* __restrict__ attn)                   // out [B,64]
{
  const int lane = threadIdx.x & 31;
  const int wv   = threadIdx.x >> 5;
  const int rowBase = blockIdx.x * 128 + wv * 16;
  const int lh  = lane >> 4;
  const int l15 = lane & 15;

  const unsigned short* Abase = H + (size_t)(rowBase + l15) * lda + lh * 8;
  const unsigned short* Wbase = Wsup + (size_t)l15 * FEAT + lh * 8;

  v8f acc[4];
#pragma unroll
  for (int tn = 0; tn < 4; ++tn)
#pragma unroll
    for (int r = 0; r < 8; ++r) acc[tn][r] = 0.0f;

  bf16x16 a0, b0[4], a1, b1[4];
  auto loadAll = [&](bf16x16& aF, bf16x16* bF, int k) {
    aF = load_frag(Abase + k, 16);
#pragma unroll
    for (int tn = 0; tn < 4; ++tn)
      bF[tn] = load_frag(Wbase + (size_t)tn * 16 * FEAT + k, 16);
  };
  auto mma = [&](const bf16x16& aF, const bf16x16* bF) {
#pragma unroll
    for (int tn = 0; tn < 4; ++tn)
      acc[tn] = __builtin_amdgcn_wmma_f32_16x16x32_bf16(
          false, aF, false, bF[tn], (short)0, acc[tn], false, false);
  };

  loadAll(a0, b0, 0);
  int k = 32;
  for (; k + 32 < FEAT; k += 64) {
    loadAll(a1, b1, k);
    mma(a0, b0);
    loadAll(a0, b0, k + 32);
    mma(a1, b1);
  }
  loadAll(a1, b1, FEAT - 32);
  mma(a0, b0);
  mma(a1, b1);

  // Row softmax: each row's 64 values live in {4 accs} x {16 lanes of a half}
#pragma unroll
  for (int r = 0; r < 8; ++r) {
    float m = acc[0][r];
#pragma unroll
    for (int tn = 1; tn < 4; ++tn) m = fmaxf(m, acc[tn][r]);
#pragma unroll
    for (int mask = 1; mask <= 8; mask <<= 1)
      m = fmaxf(m, __shfl_xor(m, mask, 32));
    float e[4], s = 0.0f;
#pragma unroll
    for (int tn = 0; tn < 4; ++tn) { e[tn] = __expf(acc[tn][r] - m); s += e[tn]; }
#pragma unroll
    for (int mask = 1; mask <= 8; mask <<= 1)
      s += __shfl_xor(s, mask, 32);
    const float inv = 1.0f / s;
    const int row = rowBase + lh * 8 + r;
#pragma unroll
    for (int tn = 0; tn < 4; ++tn)
      attn[(size_t)row * SS + tn * 16 + l15] = f2bf(e[tn] * inv);
  }
}

// ---------------------------------------------------------------------------
// Elementwise LSTM update. Z already contains G0 + biases + GEMM result.
// Gate order i,f,g,o. Writes c (ws), h f32 (d_out), h bf16 (X right half).
// ---------------------------------------------------------------------------
__global__ __launch_bounds__(256) void lstm_update_kernel(
    const float* __restrict__ Z, const float* __restrict__ q,
    float* __restrict__ c, float* __restrict__ hout,
    unsigned short* __restrict__ X)
{
  const int idx = blockIdx.x * 256 + threadIdx.x;     // 0 .. B*F-1
  const int b = idx >> 10;
  const int f = idx & 1023;
  const size_t g = (size_t)b * 4096;
  const float gi = Z[g + f];
  const float gf = Z[g + 1024 + f];
  const float gg = Z[g + 2048 + f];
  const float go = Z[g + 3072 + f];
  const float cn = fast_sig(gf) * c[idx] + fast_sig(gi) * fast_tanh(gg);
  const float hn = fast_sig(go) * fast_tanh(cn) + q[idx];
  c[idx] = cn;
  hout[idx] = hn;
  X[(size_t)b * 2048 + 1024 + f] = f2bf(hn);
}

// --------------------------- setup kernels ---------------------------------
__global__ void build_wihl_kernel(const float* __restrict__ W_ih,   // [4096,2048]
                                  unsigned short* __restrict__ out) { // [4096,1024]
  int i = blockIdx.x * 256 + threadIdx.x;                // 0..4096*1024-1
  int n = i >> 10, kk = i & 1023;
  out[i] = f2bf(W_ih[(size_t)n * 2048 + kk]);
}
__global__ void build_wcat_kernel(const float* __restrict__ W_ih,   // [4096,2048]
                                  const float* __restrict__ W_hh,   // [4096,1024]
                                  unsigned short* __restrict__ out) { // [4096,2048]
  int i = blockIdx.x * 256 + threadIdx.x;                // 0..4096*2048-1
  int n = i >> 11, j = i & 2047;
  float v = (j < 1024) ? W_ih[(size_t)n * 2048 + 1024 + j]
                       : W_hh[(size_t)n * 1024 + (j - 1024)];
  out[i] = f2bf(v);
}
__global__ void f32_to_bf16_kernel(const float* __restrict__ in,
                                   unsigned short* __restrict__ out, int n) {
  int i = blockIdx.x * 256 + threadIdx.x;
  if (i < n) out[i] = f2bf(in[i]);
}
__global__ void transpose_supp_kernel(const float* __restrict__ in,   // [64,F]
                                      unsigned short* __restrict__ out) { // [F,64]
  int i = blockIdx.x * 256 + threadIdx.x;      // 0..F*64-1
  int f = i >> 6, s = i & 63;
  out[i] = f2bf(in[(size_t)s * FEAT + f]);
}
__global__ void query_to_X_kernel(const float* __restrict__ q,
                                  unsigned short* __restrict__ X) {
  int i = blockIdx.x * 256 + threadIdx.x;      // 0..B*F-1
  int b = i >> 10, f = i & 1023;
  X[(size_t)b * 2048 + 1024 + f] = f2bf(q[i]);
}
__global__ void zero_f32_kernel(float* __restrict__ p, int n) {
  int i = blockIdx.x * 256 + threadIdx.x;
  if (i < n) p[i] = 0.0f;
}
__global__ void add_bias_kernel(float* __restrict__ G0,
                                const float* __restrict__ bih,
                                const float* __restrict__ bhh) {
  int i = blockIdx.x * 256 + threadIdx.x;      // 0..B*4096-1
  int j = i & 4095;
  G0[i] += bih[j] + bhh[j];
}

// ---------------------------------------------------------------------------
extern "C" void kernel_launch(void* const* d_in, const int* in_sizes, int n_in,
                              void* d_out, int out_size, void* d_ws, size_t ws_size,
                              hipStream_t stream) {
  (void)in_sizes; (void)n_in; (void)out_size;
  const float* query   = (const float*)d_in[0];
  const float* support = (const float*)d_in[1];
  const float* W_ih    = (const float*)d_in[2];   // [4F, 2F]
  const float* W_hh    = (const float*)d_in[3];   // [4F, F]
  const float* b_ih    = (const float*)d_in[4];
  const float* b_hh    = (const float*)d_in[5];
  float* hout = (float*)d_out;

  char* ws = (char*)d_ws;
  size_t off = 0;
  auto take = [&](size_t bytes) -> char* {
    char* p = ws + off;
    off = (off + bytes + 255) & ~(size_t)255;
    return p;
  };
  unsigned short* X     = (unsigned short*)take((size_t)Bsz * 2048 * 2);  // [readout|h]
  unsigned short* WihL  = (unsigned short*)take((size_t)4096 * 1024 * 2); // W_ih left, bf16
  unsigned short* Wcat  = (unsigned short*)take((size_t)4096 * 2048 * 2); // [W_ihR|W_hh]
  unsigned short* suppB = (unsigned short*)take((size_t)SS * FEAT * 2);
  unsigned short* suppT = (unsigned short*)take((size_t)FEAT * SS * 2);
  unsigned short* attnB = (unsigned short*)take((size_t)Bsz * SS * 2);
  float*          Zbuf  = (float*)take((size_t)Bsz * 4096 * 4);
  float*          G0    = (float*)take((size_t)Bsz * 4096 * 4);
  float*          cbuf  = (float*)take((size_t)Bsz * FEAT * 4);
  if (off > ws_size) return;  // workspace too small; cannot proceed safely

  // ---- one-time setup ----
  build_wihl_kernel<<<(4096 * 1024) / 256, 256, 0, stream>>>(W_ih, WihL);
  build_wcat_kernel<<<(4096 * 2048) / 256, 256, 0, stream>>>(W_ih, W_hh, Wcat);
  f32_to_bf16_kernel<<<(SS * FEAT) / 256, 256, 0, stream>>>(support, suppB, SS * FEAT);
  transpose_supp_kernel<<<(FEAT * SS) / 256, 256, 0, stream>>>(support, suppT);
  query_to_X_kernel<<<(Bsz * FEAT) / 256, 256, 0, stream>>>(query, X);   // h0 = query
  zero_f32_kernel<<<(Bsz * FEAT) / 256, 256, 0, stream>>>(cbuf, Bsz * FEAT);

  // G0 = query @ W_ih[:, :F]^T, then fold in b_ih + b_hh (constant across steps)
  wmma_gemm<0, 0><<<dim3(4096 / 256, Bsz / 128), 256, 0, stream>>>(
      X + 1024, 2048, WihL, 1024, nullptr, (void*)G0, 4096, FEAT);
  add_bias_kernel<<<(Bsz * 4096) / 256, 256, 0, stream>>>(G0, b_ih, b_hh);

  // ---- 64 sequential steps ----
  for (int t = 0; t < SS; ++t) {
    // attn = softmax(h @ support^T)              [B,64] bf16
    attn_softmax_kernel<<<Bsz / 128, 256, 0, stream>>>(X + 1024, 2048, suppB, attnB);
    // readout = attn @ support -> X[:, :F]       (bf16 out, ldc = 2048)
    wmma_gemm<0, 1><<<dim3(FEAT / 256, Bsz / 128), 256, 0, stream>>>(
        attnB, SS, suppT, SS, nullptr, (void*)X, 2048, SS);
    // Z = G0 + [readout|h] @ [W_ihR|W_hh]^T      (single fused K=2048 GEMM)
    wmma_gemm<1, 0><<<dim3(4096 / 256, Bsz / 128), 256, 0, stream>>>(
        X, 2048, Wcat, 2048, G0, (void*)Zbuf, 4096, 2048);
    // gate math; writes h -> d_out (f32), X right half (bf16), c -> ws
    lstm_update_kernel<<<(Bsz * FEAT) / 256, 256, 0, stream>>>(
        Zbuf, query, cbuf, hout, X);
  }
}
